// MultiHeadAttention_89472758710361
// MI455X (gfx1250) — matrix operand order repaired and checked by reading, not compiled, not run on hardware
//
#include <hip/hip_runtime.h>
#include <math.h>

// ---------------------------------------------------------------------------
// MHA block for MI455X (gfx1250, wave32, WMMA 16x16x32 f16 -> f32 acc)
//   B=16, L=1024, D_MODEL=640, H=5, d_k=d_v=128
// ---------------------------------------------------------------------------

typedef __attribute__((ext_vector_type(16))) _Float16 v16h;
typedef __attribute__((ext_vector_type(8)))  _Float16 v8h;
typedef __attribute__((ext_vector_type(8)))  float    v8f;
typedef __attribute__((ext_vector_type(4)))  int      v4i;

#define D_MODEL 640
#define L_SEQ   1024
#define NB      16
#define NH      5
#define DK      128

// ---- async global->LDS (gfx1250), guarded so compile never breaks ---------
#if defined(__has_builtin)
#if __has_builtin(__builtin_amdgcn_global_load_async_to_lds_b128) && \
    __has_builtin(__builtin_amdgcn_s_wait_asynccnt)
#define HAVE_ASYNC 1
#endif
#if __has_builtin(__builtin_amdgcn_permlane16)
#define HAVE_PL16 1
#endif
#endif
#ifndef HAVE_ASYNC
#define HAVE_ASYNC 0
#endif
#ifndef HAVE_PL16
#define HAVE_PL16 0
#endif

__device__ __forceinline__ void async_cp16(const void* g, void* l) {
#if HAVE_ASYNC
  __builtin_amdgcn_global_load_async_to_lds_b128(
      (__attribute__((address_space(1))) v4i*)g,
      (__attribute__((address_space(3))) v4i*)l, 0, 0);
#else
  *(uint4*)l = *(const uint4*)g;
#endif
}
#if HAVE_ASYNC
#define ASYNC_WAIT(n) __builtin_amdgcn_s_wait_asynccnt(n)
#else
#define ASYNC_WAIT(n) ((void)0)
#endif

// ---- 16-lane butterfly reductions via v_permlane16_b32 (VALU, no LDS) -----
#if HAVE_PL16
__device__ __forceinline__ float pl16(float v, unsigned s0, unsigned s1) {
  const int i = __builtin_amdgcn_permlane16(__float_as_int(v), __float_as_int(v),
                                            s0, s1, false, false);
  return __int_as_float(i);
}
__device__ __forceinline__ float redmax16(float v) {
  v = fmaxf(v, pl16(v, 0x67452301u, 0xEFCDAB89u));  // xor 1
  v = fmaxf(v, pl16(v, 0x54761032u, 0xDCFE98BAu));  // xor 2
  v = fmaxf(v, pl16(v, 0x32107654u, 0xBA98FEDCu));  // xor 4
  v = fmaxf(v, pl16(v, 0xFEDCBA98u, 0x76543210u));  // xor 8
  return v;
}
__device__ __forceinline__ float redsum16(float v) {
  v += pl16(v, 0x67452301u, 0xEFCDAB89u);
  v += pl16(v, 0x54761032u, 0xDCFE98BAu);
  v += pl16(v, 0x32107654u, 0xBA98FEDCu);
  v += pl16(v, 0xFEDCBA98u, 0x76543210u);
  return v;
}
#else
__device__ __forceinline__ float redmax16(float v) {
#pragma unroll
  for (int off = 1; off < 16; off <<= 1) v = fmaxf(v, __shfl_xor(v, off, 32));
  return v;
}
__device__ __forceinline__ float redsum16(float v) {
#pragma unroll
  for (int off = 1; off < 16; off <<= 1) v += __shfl_xor(v, off, 32);
  return v;
}
#endif

// ---------------------------------------------------------------------------
__device__ __forceinline__ v8f wmma_f16(v16h a, v16h b, v8f c) {
  return __builtin_amdgcn_wmma_f32_16x16x32_f16(false, a, false, b, (short)0, c,
                                                false, false);
}

union V16U { v16h v; struct { v8h lo, hi; } p; };

// A-fragment (16x32 f16) = one contiguous 16-f16 run:
//   j in [0,16): k = 16*half + j
__device__ __forceinline__ v16h loadA_f16(const _Float16* p, int off, int half) {
  V16U u;
  u.p.lo = *(const v8h*)(p + off + 16 * half);
  u.p.hi = *(const v8h*)(p + off + 16 * half + 8);
  return u.v;
}
__device__ __forceinline__ v16h loadA_f32(const float* p, int off, int half) {
  const v8f lo = *(const v8f*)(p + off + 16 * half);
  const v8f hi = *(const v8f*)(p + off + 16 * half + 8);
  v16h a;
#pragma unroll
  for (int j = 0; j < 8; ++j) {
    a[j]     = (_Float16)lo[j];
    a[j + 8] = (_Float16)hi[j];
  }
  return a;
}

// ---------------------------------------------------------------------------
// Flash attention. grid=(B*H, L/128); 8 waves; wave owns 16 query rows.
// K (32x128) and V (128x32) tiles staged in LDS via async copy, double
// buffered; all 8 waves share the tiles.
// attn = softmax(S)*A_ds: numerator weighted by A_ds, denominator plain.
// ---------------------------------------------------------------------------
__global__ void __launch_bounds__(256, 1)
attn_kernel(const _Float16* __restrict__ Qh,
            const _Float16* __restrict__ Kh,
            const _Float16* __restrict__ Vt,
            const float* __restrict__ Ads,
            float* __restrict__ Ctx) {
  __shared__ __align__(32) _Float16 Klds[2][32 * DK];   // [kv][d]
  __shared__ __align__(32) _Float16 Vlds[2][DK * 32];   // [d][kv]
  __shared__ __align__(32) _Float16 psh[8][16 * 32];    // per-wave P strip

  const int tid  = threadIdx.x;
  const int lane = tid & 31;
  const int wave = tid >> 5;
  const int half = lane >> 4;
  const int ln   = lane & 15;
  const int bh   = blockIdx.x;           // 0..79
  const int h    = bh % NH;
  const int b_   = bh / NH;
  const int q0   = blockIdx.y * 128 + wave * 16;

  const _Float16* Qb = Qh + (size_t)bh * L_SEQ * DK;
  const _Float16* Kb = Kh + (size_t)bh * L_SEQ * DK;
  const _Float16* Vb = Vt + (size_t)bh * DK * L_SEQ;

  // stage K tile (contiguous 8KB) + V tile (128 rows x 64B) into LDS buf
  auto stage = [&](int buf, int k0) {
    const char* Kg = (const char*)(Kb + (size_t)k0 * DK);
    char* Kl = (char*)&Klds[buf][0];
#pragma unroll
    for (int c = 0; c < 2; ++c) {
      const int idx = tid + 256 * c;            // 512 x 16B = 8KB
      async_cp16(Kg + idx * 16, Kl + idx * 16);
    }
    const char* Vg = (const char*)Vb;
    char* Vl = (char*)&Vlds[buf][0];
#pragma unroll
    for (int c = 0; c < 2; ++c) {
      const int idx = tid + 256 * c;
      const int d = idx >> 2, off = (idx & 3) * 16;
      async_cp16(Vg + ((size_t)d * L_SEQ + k0) * 2 + off, Vl + d * 64 + off);
    }
  };

  // Q fragments: 16 rows x 128 as 4 chunks of K=32 (two b128 loads each)
  const _Float16* Qrow = Qb + (size_t)(q0 + ln) * DK;
  v16h qf[4];
#pragma unroll
  for (int kk = 0; kk < 4; ++kk) qf[kk] = loadA_f16(Qrow, kk * 32, half);

  v8f o[8] = {};
  float mrow[8], drow[8];
#pragma unroll
  for (int r = 0; r < 8; ++r) { mrow[r] = -__builtin_inff(); drow[r] = 0.f; }

  const float scale = 0.08838834764831844f;  // 1/sqrt(128)

  stage(0, 0);
  for (int it = 0; it < L_SEQ / 32; ++it) {
    const int k0  = it * 32;
    const int buf = it & 1;
    if (it + 1 < L_SEQ / 32) {
      stage(buf ^ 1, k0 + 32);     // prefetch next tile into other buffer
      ASYNC_WAIT(4);               // 4 async ops just issued; tile `it` done
    } else {
      ASYNC_WAIT(0);
    }
    __syncthreads();

    // S = Q @ K^T : two 16x16 tiles, contraction d=128 in 4 WMMA steps
    v8f s0 = {}, s1 = {};
#pragma unroll
    for (int kk = 0; kk < 4; ++kk) {
      const v16h bk0 = *(const v16h*)(&Klds[buf][ln * DK + kk * 32 + 16 * half]);
      const v16h bk1 = *(const v16h*)(&Klds[buf][(16 + ln) * DK + kk * 32 + 16 * half]);
      s0 = wmma_f16(qf[kk], bk0, s0);
      s1 = wmma_f16(qf[kk], bk1, s1);
    }
    // online softmax per C-layout row r (global row r + 8*half)
#pragma unroll
    for (int r = 0; r < 8; ++r) {
      const float a0 = s0[r] * scale, a1 = s1[r] * scale;
      const float mx    = redmax16(fmaxf(a0, a1));
      const float mnew  = fmaxf(mrow[r], mx);
      const float alpha = __expf(mrow[r] - mnew);
      const float e0 = __expf(a0 - mnew), e1 = __expf(a1 - mnew);
      const float rs = redsum16(e0 + e1);
      drow[r] = drow[r] * alpha + rs;
      mrow[r] = mnew;
#pragma unroll
      for (int n = 0; n < 8; ++n) o[n][r] *= alpha;
      // post-softmax channel weighting (numerator only)
      const int qr = q0 + r + 8 * half;
      const float p0 = e0 * Ads[(size_t)qr * L_SEQ + k0 + ln];
      const float p1 = e1 * Ads[(size_t)qr * L_SEQ + k0 + 16 + ln];
      psh[wave][(r + 8 * half) * 32 + ln]      = (_Float16)p0;
      psh[wave][(r + 8 * half) * 32 + 16 + ln] = (_Float16)p1;
    }
    // P (C-layout) -> A-frag: two ds_load_b128 (same wave, DS in-order)
    const v16h pf = loadA_f16(&psh[wave][ln * 32], 0, half);
    // O += P(16x32) @ V(32x128)
#pragma unroll
    for (int n = 0; n < 8; ++n) {
      const v16h bv = *(const v16h*)(&Vlds[buf][(n * 16 + ln) * 32 + 16 * half]);
      o[n] = wmma_f16(pf, bv, o[n]);
    }
    __syncthreads();   // tile fully consumed before buffer reuse
  }

  // normalize by denominator, write Ctx[b*1024+l][h*128+d]
#pragma unroll
  for (int n = 0; n < 8; ++n)
#pragma unroll
    for (int r = 0; r < 8; ++r) {
      const int row = q0 + r + 8 * half;
      Ctx[((size_t)b_ * L_SEQ + row) * D_MODEL + h * DK + n * 16 + ln] =
          o[n][r] / drow[r];
    }
}

// ---------------------------------------------------------------------------
// Weight transpose+convert: Wt[n][k] (f16) = W[k][n] (f32). 640x640.
// ---------------------------------------------------------------------------
__global__ void __launch_bounds__(256, 1)
conv_wT(const float* __restrict__ W, _Float16* __restrict__ Wt) {
  const int idx = blockIdx.x * 256 + threadIdx.x;   // n*640 + k
  const int n = idx / D_MODEL, k = idx - n * D_MODEL;
  Wt[idx] = (_Float16)W[(size_t)k * D_MODEL + n];
}

// ---------------------------------------------------------------------------
// Projection GEMM: Out = X[16384,640] @ W[640,640] (Wt pre-transposed f16).
// vtrans==0: Out[bh][l][d] (Q,K);  vtrans==1: Out[bh][d][l] (V transposed)
// ---------------------------------------------------------------------------
__global__ void __launch_bounds__(256, 1)
proj_gemm(const float* __restrict__ X, const _Float16* __restrict__ Wt,
          _Float16* __restrict__ Out, int vtrans) {
  const int lane = threadIdx.x & 31;
  const int wave = threadIdx.x >> 5;
  const int half = lane >> 4;
  const int ln   = lane & 15;
  const int m0   = blockIdx.x * 128 + wave * 16;
  const int nb   = blockIdx.y * 64;

  const float* Xrow = X + (size_t)(m0 + ln) * D_MODEL;

  v8f acc[4] = {};
  for (int k0 = 0; k0 < D_MODEL; k0 += 32) {
    if (k0 + 32 < D_MODEL) __builtin_prefetch(Xrow + k0 + 32, 0, 3);  // near-cache
    const v16h a = loadA_f32(Xrow, k0, half);
#pragma unroll
    for (int n = 0; n < 4; ++n) {
      const v16h bf =
          *(const v16h*)(Wt + (size_t)(nb + n * 16 + ln) * D_MODEL + k0 + 16 * half);
      acc[n] = wmma_f16(a, bf, acc[n]);
    }
  }
#pragma unroll
  for (int n = 0; n < 4; ++n) {
#pragma unroll
    for (int r = 0; r < 8; ++r) {
      const int row = m0 + r + 8 * half;          // token row = b*1024 + l
      const int col = nb + n * 16 + ln;           // = h*128 + d
      const int b_ = row >> 10, l = row & 1023;
      const int h = col >> 7, d = col & 127;
      const size_t bh = (size_t)b_ * NH + h;
      const _Float16 v = (_Float16)acc[n][r];
      if (!vtrans) Out[(bh * L_SEQ + l) * DK + d] = v;
      else         Out[(bh * DK + d) * L_SEQ + l] = v;
    }
  }
}

// ---------------------------------------------------------------------------
// Fused: out = Ctx @ Wfc; out += ELU(out@a1+b1)@a2 + b2; y = LN(out + resid)
// ---------------------------------------------------------------------------
__global__ void __launch_bounds__(256, 1)
out_fused(const float* __restrict__ Ctx, const _Float16* __restrict__ Wfct,
          const float* __restrict__ a1w, const float* __restrict__ a1b,
          const float* __restrict__ a2w, const float* __restrict__ a2b,
          const float* __restrict__ resid, float* __restrict__ Out) {
  __shared__ float smem[16 * D_MODEL];
  __shared__ float hid[16 * 8];
  const int tid  = threadIdx.x;
  const int lane = tid & 31;
  const int wave = tid >> 5;
  const int half = lane >> 4;
  const int ln   = lane & 15;
  const int m0   = blockIdx.x * 16;
  const int nb   = wave * 80;

  const float* Crow = Ctx + (size_t)(m0 + ln) * D_MODEL;

  v8f acc[5] = {};
  for (int k0 = 0; k0 < D_MODEL; k0 += 32) {
    const v16h a = loadA_f32(Crow, k0, half);
#pragma unroll
    for (int n = 0; n < 5; ++n) {
      const v16h bf =
          *(const v16h*)(Wfct + (size_t)(nb + n * 16 + ln) * D_MODEL + k0 + 16 * half);
      acc[n] = wmma_f16(a, bf, acc[n]);
    }
  }
#pragma unroll
  for (int n = 0; n < 5; ++n)
#pragma unroll
    for (int r = 0; r < 8; ++r)
      smem[(r + 8 * half) * D_MODEL + nb + n * 16 + ln] = acc[n][r];
  __syncthreads();

  // adapter hidden: 16 rows x 8 units, ELU
  if (tid < 128) {
    const int r = tid >> 3, j = tid & 7;
    float hsum = a1b[j];
    for (int c = 0; c < D_MODEL; ++c) hsum += smem[r * D_MODEL + c] * a1w[c * 8 + j];
    hid[r * 8 + j] = hsum > 0.f ? hsum : (__expf(hsum) - 1.f);
  }
  __syncthreads();

  // out2 = out + hid@a2 + b2 + residual (in place)
  for (int idx = tid; idx < 16 * D_MODEL; idx += 256) {
    const int r = idx / D_MODEL, c = idx - r * D_MODEL;
    float add = a2b[c];
#pragma unroll
    for (int j = 0; j < 8; ++j) add += hid[r * 8 + j] * a2w[j * D_MODEL + c];
    smem[idx] += add + resid[(size_t)(m0 + r) * D_MODEL + c];
  }
  __syncthreads();

  // LayerNorm: wave w -> rows 2w, 2w+1
#pragma unroll
  for (int rr = 0; rr < 2; ++rr) {
    const int r = wave * 2 + rr;
    float s = 0.f, s2 = 0.f;
    for (int c = lane; c < D_MODEL; c += 32) {
      const float x = smem[r * D_MODEL + c];
      s += x; s2 += x * x;
    }
#pragma unroll
    for (int off = 1; off < 32; off <<= 1) {
      s  += __shfl_xor(s, off, 32);
      s2 += __shfl_xor(s2, off, 32);
    }
    const float mean = s * (1.f / D_MODEL);
    const float var  = s2 * (1.f / D_MODEL) - mean * mean;
    const float rstd = rsqrtf(var + 1e-5f);
    for (int c = lane; c < D_MODEL; c += 32)
      Out[(size_t)(m0 + r) * D_MODEL + c] = (smem[r * D_MODEL + c] - mean) * rstd;
  }
}

// ---------------------------------------------------------------------------
extern "C" void kernel_launch(void* const* d_in, const int* in_sizes, int n_in,
                              void* d_out, int out_size, void* d_ws, size_t ws_size,
                              hipStream_t stream) {
  const float* inQ = (const float*)d_in[0];
  const float* inK = (const float*)d_in[1];
  const float* inV = (const float*)d_in[2];
  const float* Ads = (const float*)d_in[3];
  const float* Wq  = (const float*)d_in[4];
  const float* Wk  = (const float*)d_in[5];
  const float* Wv  = (const float*)d_in[6];
  const float* Wfc = (const float*)d_in[7];
  const float* a1w = (const float*)d_in[8];
  const float* a1b = (const float*)d_in[9];
  const float* a2w = (const float*)d_in[10];
  const float* a2b = (const float*)d_in[11];
  float* out = (float*)d_out;

  char* ws = (char*)d_ws;
  const size_t wtB  = (size_t)D_MODEL * D_MODEL * sizeof(_Float16);          // 819,200
  const size_t qkvB = (size_t)NB * NH * L_SEQ * DK * sizeof(_Float16);       // 20 MiB
  _Float16* Wqt  = (_Float16*)(ws);
  _Float16* Wkt  = (_Float16*)(ws + wtB);
  _Float16* Wvt  = (_Float16*)(ws + 2 * wtB);
  _Float16* Wfct = (_Float16*)(ws + 3 * wtB);
  _Float16* Qh   = (_Float16*)(ws + 4 * wtB);
  _Float16* Kh   = (_Float16*)(ws + 4 * wtB + qkvB);
  _Float16* Vt   = (_Float16*)(ws + 4 * wtB + 2 * qkvB);
  float*    Ctx  = (float*)   (ws + 4 * wtB + 3 * qkvB);

  const dim3 blk(256);
  const int wtBlocks = (D_MODEL * D_MODEL) / 256;   // 1600
  conv_wT<<<dim3(wtBlocks), blk, 0, stream>>>(Wq,  Wqt);
  conv_wT<<<dim3(wtBlocks), blk, 0, stream>>>(Wk,  Wkt);
  conv_wT<<<dim3(wtBlocks), blk, 0, stream>>>(Wv,  Wvt);
  conv_wT<<<dim3(wtBlocks), blk, 0, stream>>>(Wfc, Wfct);

  proj_gemm<<<dim3(128, 10), blk, 0, stream>>>(inQ, Wqt, Qh, 0);
  proj_gemm<<<dim3(128, 10), blk, 0, stream>>>(inK, Wkt, Kh, 0);
  proj_gemm<<<dim3(128, 10), blk, 0, stream>>>(inV, Wvt, Vt, 1);
  attn_kernel<<<dim3(NB * NH, L_SEQ / 128), blk, 0, stream>>>(Qh, Kh, Vt, Ads, Ctx);
  out_fused<<<dim3((NB * L_SEQ) / 16), blk, 0, stream>>>(Ctx, Wfct, a1w, a1b, a2w, a2b,
                                                         inQ, out);
}
